// SpectralStateSpaceBlock_5798205849735
// MI455X (gfx1250) — matrix operand order repaired and checked by reading, not compiled, not run on hardware
//
#include <hip/hip_runtime.h>
#include <math.h>

#define BB 2
#define SS 2048
#define DD 1024
#define FF 256
#define HH 16
#define HDIM 64
#define KCODE 1024
#define NTOK 4096   /* B*S */
#define D4 4096
#define F2 512

typedef __attribute__((ext_vector_type(16))) __bf16       bf16x16;
typedef __attribute__((ext_vector_type(8)))  float        floatx8;
typedef __attribute__((ext_vector_type(4)))  unsigned int uintx4;

union FragBF { bf16x16 v; uintx4 q[2]; };

__device__ __forceinline__ unsigned short f32_to_bf16(float f) {
  unsigned int u = __float_as_uint(f);
  u += 0x7fffu + ((u >> 16) & 1u);          // round-to-nearest-even
  return (unsigned short)(u >> 16);
}

__device__ __forceinline__ floatx8 wmma_bf16(FragBF a, FragBF b, floatx8 c) {
  return __builtin_amdgcn_wmma_f32_16x16x32_bf16(false, a.v, false, b.v,
                                                 (short)0, c, false, false);
}

// Low 32 bits of a generic pointer to __shared__ == wave-relative LDS byte
// address (flat LDS aperture is {aperture_hi, lds_offset}).
__device__ __forceinline__ unsigned lds_addr32(const void* p) {
  return (unsigned)(unsigned long long)p;
}

// ---------------------------------------------------------------------------
// f32 -> bf16 conversion
// ---------------------------------------------------------------------------
__global__ void cvt_bf16_kernel(const float* __restrict__ a,
                                unsigned short* __restrict__ o, size_t n) {
  size_t i = (size_t)blockIdx.x * 256 + threadIdx.x;
  if (i < n) o[i] = f32_to_bf16(a[i]);
}

// ---------------------------------------------------------------------------
// LayerNorm: one block per row of D=1024; writes fp32 and/or bf16
// ---------------------------------------------------------------------------
__global__ __launch_bounds__(256) void ln_kernel(const float* __restrict__ x,
    const float* __restrict__ w, const float* __restrict__ b,
    float* __restrict__ outF, unsigned short* __restrict__ outB) {
  __shared__ float s1[256], s2[256];
  int row = blockIdx.x, tid = threadIdx.x;
  const float* xr = x + (size_t)row * DD;
  float v[4], lsum = 0.f, lsq = 0.f;
#pragma unroll
  for (int i = 0; i < 4; ++i) {
    v[i] = xr[tid + 256 * i];
    lsum += v[i]; lsq += v[i] * v[i];
  }
  s1[tid] = lsum; s2[tid] = lsq; __syncthreads();
  for (int s = 128; s > 0; s >>= 1) {
    if (tid < s) { s1[tid] += s1[tid + s]; s2[tid] += s2[tid + s]; }
    __syncthreads();
  }
  float mean = s1[0] * (1.f / DD);
  float var  = s2[0] * (1.f / DD) - mean * mean;
  float rstd = rsqrtf(var + 1e-5f);
#pragma unroll
  for (int i = 0; i < 4; ++i) {
    int c = tid + 256 * i;
    float y = (v[i] - mean) * rstd * w[c] + b[c];
    if (outF) outF[(size_t)row * DD + c] = y;
    if (outB) outB[(size_t)row * DD + c] = f32_to_bf16(y);
  }
}

// ---------------------------------------------------------------------------
// bf16 WMMA GEMM:  C[M,N] = act( A[M,K] @ W[N,K]^T + bias )
// block = 256 threads (8 waves), 128x128 tile, K-chunks of 32.
// Double-buffered LDS tiles filled with GLOBAL_LOAD_ASYNC_TO_LDS_B128
// (ASYNCcnt-tracked; async loads retire in order, so s_wait_asynccnt 4
// guarantees the previous buffer's 4 ops/wave are complete). The last
// K-chunk is peeled so the steady-state loop is branch-free.
// ---------------------------------------------------------------------------
#define ACT_NONE 0
#define ACT_SIGMOID 1
#define ACT_GELU 2
#define LSTR 48  /* padded LDS row stride in bf16 elements (96B, 16B aligned) */

__global__ __launch_bounds__(256) void gemm_bf16_kernel(
    const unsigned short* __restrict__ A, const unsigned short* __restrict__ W,
    const float* __restrict__ bias, float* __restrict__ outF,
    unsigned short* __restrict__ outB, int M, int N, int K, int act) {
  __shared__ __align__(16) unsigned short lA[2][128 * LSTR];
  __shared__ __align__(16) unsigned short lB[2][128 * LSTR];
  int tid = threadIdx.x;
  int lane = tid & 31, wid = tid >> 5;
  int l15 = lane & 15, half = lane >> 4;
  int bm = blockIdx.y * 128, bn = blockIdx.x * 128;
  int wm = (wid & 1) * 64;     // 4 x 16-row subtiles
  int wn = (wid >> 1) * 32;    // 2 x 16-col subtiles

  // per-thread tile segment: 2 x 16B per tile (512 segments of 8 bf16)
  int seg0r = tid >> 2,          seg0c = (tid & 3) * 8;
  int seg1r = (tid + 256) >> 2,  seg1c = ((tid + 256) & 3) * 8;
  unsigned l0 = (unsigned)((seg0r * LSTR + seg0c) * 2);
  unsigned l1 = (unsigned)((seg1r * LSTR + seg1c) * 2);

  auto issue = [&](int buf, int k0) {
    unsigned la = lds_addr32(&lA[buf][0]);
    unsigned lb = lds_addr32(&lB[buf][0]);
    const unsigned short* ga0 = &A[(size_t)(bm + seg0r) * K + k0 + seg0c];
    const unsigned short* gw0 = &W[(size_t)(bn + seg0r) * K + k0 + seg0c];
    const unsigned short* ga1 = &A[(size_t)(bm + seg1r) * K + k0 + seg1c];
    const unsigned short* gw1 = &W[(size_t)(bn + seg1r) * K + k0 + seg1c];
    asm volatile("global_load_async_to_lds_b128 %0, %1, off"
                 :: "v"(la + l0), "v"(ga0) : "memory");
    asm volatile("global_load_async_to_lds_b128 %0, %1, off"
                 :: "v"(lb + l0), "v"(gw0) : "memory");
    asm volatile("global_load_async_to_lds_b128 %0, %1, off"
                 :: "v"(la + l1), "v"(ga1) : "memory");
    asm volatile("global_load_async_to_lds_b128 %0, %1, off"
                 :: "v"(lb + l1), "v"(gw1) : "memory");
  };

  floatx8 acc[4][2];
#pragma unroll
  for (int mt = 0; mt < 4; ++mt)
#pragma unroll
    for (int nt = 0; nt < 2; ++nt)
#pragma unroll
      for (int r = 0; r < 8; ++r) acc[mt][nt][r] = 0.f;

  auto compute = [&](int cur) {
    const unsigned short* cA = lA[cur];
    const unsigned short* cB = lB[cur];
    FragBF bfr[2], afr[4];
#pragma unroll
    for (int nt = 0; nt < 2; ++nt) {
      int n = wn + nt * 16 + l15;
      bfr[nt].q[0] = *(const uintx4*)&cB[n * LSTR + half * 8];
      bfr[nt].q[1] = *(const uintx4*)&cB[n * LSTR + 16 + half * 8];
    }
#pragma unroll
    for (int mt = 0; mt < 4; ++mt) {
      int m = wm + mt * 16 + l15;
      afr[mt].q[0] = *(const uintx4*)&cA[m * LSTR + half * 8];
      afr[mt].q[1] = *(const uintx4*)&cA[m * LSTR + 16 + half * 8];
    }
#pragma unroll
    for (int mt = 0; mt < 4; ++mt)
#pragma unroll
      for (int nt = 0; nt < 2; ++nt)
        acc[mt][nt] = wmma_bf16(afr[mt], bfr[nt], acc[mt][nt]);
  };

  int nchunks = K >> 5;
  issue(0, 0);
  for (int c = 0; c < nchunks - 1; ++c) {
    issue((c + 1) & 1, (c + 1) << 5);
    asm volatile("s_wait_asynccnt 0x4" ::: "memory");
    __syncthreads();
    compute(c & 1);
    __syncthreads();
  }
  asm volatile("s_wait_asynccnt 0x0" ::: "memory");
  __syncthreads();
  compute((nchunks - 1) & 1);

#pragma unroll
  for (int mt = 0; mt < 4; ++mt)
#pragma unroll
    for (int nt = 0; nt < 2; ++nt) {
      int col = bn + wn + nt * 16 + l15;
      float bv = bias ? bias[col] : 0.f;
#pragma unroll
      for (int r = 0; r < 8; ++r) {
        int row = bm + wm + mt * 16 + half * 8 + r;
        float v = acc[mt][nt][r] + bv;
        if (act == ACT_SIGMOID) v = 1.f / (1.f + __expf(-v));
        else if (act == ACT_GELU) v = 0.5f * v * (1.f + erff(v * 0.70710678f));
        size_t o = (size_t)row * N + col;
        if (outF) outF[o] = v;
        if (outB) outB[o] = f32_to_bf16(v);
      }
    }
}

// ---------------------------------------------------------------------------
// Spectral complex recurrence: z[t] = A z[t-1] + u[t]; y = rot * z
// ---------------------------------------------------------------------------
__global__ void spectral_scan_kernel(const float* __restrict__ spec,
    const float* __restrict__ log_decay, const float* __restrict__ freqs,
    float* __restrict__ out) {
  int t = blockIdx.x * 256 + threadIdx.x;
  if (t >= BB * FF) return;
  int f = t & (FF - 1), b = t >> 8;
  float decay = 1.f / (1.f + __expf(-log_decay[f]));
  float om = freqs[f] * 0.1f;
  float sr, cr; sincosf(om, &sr, &cr);
  float Ar = decay * cr, Ai = decay * sr;
  float zr = 0.f, zi = 0.f;
  const float* sp = spec + (size_t)b * SS * F2;
  float* op = out + (size_t)b * SS * F2;
  for (int s = 0; s < SS; ++s) {
    float ur = sp[(size_t)s * F2 + f];
    float ui = sp[(size_t)s * F2 + FF + f];
    float nzr = Ar * zr - Ai * zi + ur;
    float nzi = Ar * zi + Ai * zr + ui;
    zr = nzr; zi = nzi;
    op[(size_t)s * F2 + f]      = cr * zr - sr * zi;
    op[(size_t)s * F2 + FF + f] = cr * zi + sr * zr;
  }
}

// ---------------------------------------------------------------------------
// VQ helpers
// ---------------------------------------------------------------------------
__global__ __launch_bounds__(256) void emb_sq_kernel(const float* __restrict__ emb,
                                                     float* __restrict__ sq) {
  __shared__ float sh[256];
  int k = blockIdx.x, tid = threadIdx.x;
  const float* er = emb + (size_t)k * DD;
  float s = 0.f;
#pragma unroll
  for (int i = 0; i < 4; ++i) { float v = er[tid + 256 * i]; s += v * v; }
  sh[tid] = s; __syncthreads();
  for (int t = 128; t > 0; t >>= 1) { if (tid < t) sh[tid] += sh[tid + t]; __syncthreads(); }
  if (!tid) sq[k] = sh[0];
}

__global__ __launch_bounds__(256) void argmin_kernel(const float* __restrict__ dot,
    const float* __restrict__ sq, int* __restrict__ idx) {
  __shared__ float sv[256]; __shared__ int si[256];
  int m = blockIdx.x, tid = threadIdx.x;
  float best = 3.4e38f; int bi = 0;
  for (int k = tid; k < KCODE; k += 256) {
    float v = sq[k] - 2.f * dot[(size_t)m * KCODE + k];
    if (v < best) { best = v; bi = k; }
  }
  sv[tid] = best; si[tid] = bi; __syncthreads();
  for (int t = 128; t > 0; t >>= 1) {
    if (tid < t) {
      if (sv[tid + t] < sv[tid] ||
          (sv[tid + t] == sv[tid] && si[tid + t] < si[tid])) {
        sv[tid] = sv[tid + t]; si[tid] = si[tid + t];
      }
    }
    __syncthreads();
  }
  if (!tid) idx[m] = si[0];
}

__global__ __launch_bounds__(256) void vq_loss_partial_kernel(
    const float* __restrict__ xm, const float* __restrict__ emb,
    const int* __restrict__ idx, float* __restrict__ part) {
  __shared__ float sh[256];
  int m = blockIdx.x, tid = threadIdx.x;
  const float* xr = xm + (size_t)m * DD;
  const float* er = emb + (size_t)idx[m] * DD;
  float s = 0.f;
#pragma unroll
  for (int i = 0; i < 4; ++i) {
    float d = er[tid + 256 * i] - xr[tid + 256 * i];
    s += d * d;
  }
  sh[tid] = s; __syncthreads();
  for (int t = 128; t > 0; t >>= 1) { if (tid < t) sh[tid] += sh[tid + t]; __syncthreads(); }
  if (!tid) part[m] = sh[0];
}

__global__ __launch_bounds__(256) void vq_loss_final_kernel(
    const float* __restrict__ part, float* __restrict__ out) {
  __shared__ float sh[256];
  int tid = threadIdx.x; float s = 0.f;
  for (int i = tid; i < NTOK; i += 256) s += part[i];
  sh[tid] = s; __syncthreads();
  for (int t = 128; t > 0; t >>= 1) { if (tid < t) sh[tid] += sh[tid + t]; __syncthreads(); }
  if (!tid) out[0] = 1.25f * sh[0] / (float)((size_t)NTOK * DD);
}

// ---------------------------------------------------------------------------
// Elementwise kernels
// ---------------------------------------------------------------------------
__global__ void ew_mul_bf_kernel(unsigned short* __restrict__ o,
    const float* __restrict__ a, const float* __restrict__ g, size_t n) {
  size_t i = (size_t)blockIdx.x * 256 + threadIdx.x;
  if (i < n) o[i] = f32_to_bf16(a[i] * g[i]);
}
__global__ void ew_gate_add_kernel(float* __restrict__ o,
    const float* __restrict__ x, const float* __restrict__ pre,
    const float* __restrict__ g, size_t n) {
  size_t i = (size_t)blockIdx.x * 256 + threadIdx.x;
  if (i < n) o[i] = x[i] + pre[i] * g[i];
}
__global__ void ew_vq_add_kernel(float* __restrict__ o,
    const float* __restrict__ x1, const float* __restrict__ emb,
    const int* __restrict__ idx, const float* __restrict__ g, size_t n) {
  size_t i = (size_t)blockIdx.x * 256 + threadIdx.x;
  if (i >= n) return;
  size_t m = i >> 10; int c = (int)(i & (DD - 1));
  o[i] = x1[i] + emb[(size_t)idx[m] * DD + c] * g[i];
}
__global__ void ew_add_kernel(float* __restrict__ o, const float* __restrict__ a,
                              const float* __restrict__ b, size_t n) {
  size_t i = (size_t)blockIdx.x * 256 + threadIdx.x;
  if (i < n) o[i] = a[i] + b[i];
}

// ---------------------------------------------------------------------------
// RoPE on q,k + layout change to [B*H, S, 64] bf16
// ---------------------------------------------------------------------------
__global__ void rope_qk_kernel(const float* __restrict__ q,
    const float* __restrict__ k, unsigned short* __restrict__ qh,
    unsigned short* __restrict__ kh) {
  size_t i = (size_t)blockIdx.x * 256 + threadIdx.x;
  size_t total = (size_t)BB * HH * SS * 32;
  if (i >= total) return;
  int j = (int)(i & 31); size_t t = i >> 5;
  int s = (int)(t % SS); int bh = (int)(t / SS);
  int h = bh & 15, b = bh >> 4;
  float invf = __powf(10000.f, -(float)j * (1.f / 32.f));
  float sn, cs; sincosf((float)s * invf, &sn, &cs);
  size_t bi = (((size_t)b * SS + s) * HH + h) * HDIM;
  size_t bo = ((size_t)bh * SS + s) * HDIM;
  float q1 = q[bi + j], q2 = q[bi + 32 + j];
  qh[bo + j]      = f32_to_bf16(q1 * cs - q2 * sn);
  qh[bo + 32 + j] = f32_to_bf16(q2 * cs + q1 * sn);
  float k1 = k[bi + j], k2 = k[bi + 32 + j];
  kh[bo + j]      = f32_to_bf16(k1 * cs - k2 * sn);
  kh[bo + 32 + j] = f32_to_bf16(k2 * cs + k1 * sn);
}

// V -> V^T bf16 [B*H, 64, S]
__global__ void v_transpose_kernel(const float* __restrict__ v,
                                   unsigned short* __restrict__ vt) {
  size_t i = (size_t)blockIdx.x * 256 + threadIdx.x;
  size_t total = (size_t)BB * HH * SS * HDIM;
  if (i >= total) return;
  int d = (int)(i & 63); size_t t = i >> 6;
  int s = (int)(t % SS); int bh = (int)(t / SS);
  int h = bh & 15, b = bh >> 4;
  vt[((size_t)bh * HDIM + d) * SS + s] =
      f32_to_bf16(v[(((size_t)b * SS + s) * HH + h) * HDIM + d]);
}

// ---------------------------------------------------------------------------
// Flash attention: one wave per 16-row Q tile, 32-key blocks, all-WMMA.
// qh/kh: [B*H, S, 64] bf16; vt: [B*H, 64, S] bf16; out: [B*S, 1024] bf16
// ---------------------------------------------------------------------------
__device__ __forceinline__ float redmax16(float v) {
#pragma unroll
  for (int m = 1; m < 16; m <<= 1) v = fmaxf(v, __shfl_xor(v, m, 16));
  return v;
}
__device__ __forceinline__ float redsum16(float v) {
#pragma unroll
  for (int m = 1; m < 16; m <<= 1) v += __shfl_xor(v, m, 16);
  return v;
}

__global__ __launch_bounds__(256) void flash_attn_kernel(
    const unsigned short* __restrict__ qh, const unsigned short* __restrict__ kh,
    const unsigned short* __restrict__ vt, unsigned short* __restrict__ attn) {
  __shared__ __align__(16) unsigned short pls[8 * 16 * 32];
  int tid = threadIdx.x;
  int lane = tid & 31, wid = tid >> 5;
  int l15 = lane & 15, half = lane >> 4;
  unsigned short* pl = &pls[wid * 512];

  int tile = blockIdx.x * 8 + wid;      // B*H*(S/16) = 4096 tiles
  int qt = tile & 127;                  // S/16 = 128
  int bh = tile >> 7;
  int h = bh & 15, b = bh >> 4;
  int qrow = qt * 16;

  const unsigned short* Q  = qh + (size_t)bh * SS * HDIM;
  const unsigned short* Kp = kh + (size_t)bh * SS * HDIM;
  const unsigned short* Vt = vt + (size_t)bh * HDIM * SS;

  FragBF qa[2];
  {
    int m = qrow + l15;
#pragma unroll
    for (int c = 0; c < 2; ++c) {
      qa[c].q[0] = *(const uintx4*)&Q[(size_t)m * HDIM + c * 32 + half * 8];
      qa[c].q[1] = *(const uintx4*)&Q[(size_t)m * HDIM + c * 32 + 16 + half * 8];
    }
  }

  float rmax[8], rsum[8];
  floatx8 o[4];
#pragma unroll
  for (int r = 0; r < 8; ++r) { rmax[r] = -1e30f; rsum[r] = 0.f; }
#pragma unroll
  for (int t = 0; t < 4; ++t)
#pragma unroll
    for (int r = 0; r < 8; ++r) o[t][r] = 0.f;

  int qend = qrow + 15;
  for (int kb = 0; kb <= qend; kb += 32) {
    floatx8 s0, s1;
#pragma unroll
    for (int r = 0; r < 8; ++r) { s0[r] = 0.f; s1[r] = 0.f; }
#pragma unroll
    for (int c = 0; c < 2; ++c) {
      FragBF kf0, kf1;
      int key0 = kb + l15, key1 = kb + 16 + l15;
      kf0.q[0] = *(const uintx4*)&Kp[(size_t)key0 * HDIM + c * 32 + half * 8];
      kf0.q[1] = *(const uintx4*)&Kp[(size_t)key0 * HDIM + c * 32 + 16 + half * 8];
      kf1.q[0] = *(const uintx4*)&Kp[(size_t)key1 * HDIM + c * 32 + half * 8];
      kf1.q[1] = *(const uintx4*)&Kp[(size_t)key1 * HDIM + c * 32 + 16 + half * 8];
      s0 = wmma_bf16(qa[c], kf0, s0);
      s1 = wmma_bf16(qa[c], kf1, s1);
    }
    // scale + causal mask + online softmax
    int key0 = kb + l15, key1 = kb + 16 + l15;
    float mloc[8];
#pragma unroll
    for (int r = 0; r < 8; ++r) {
      int srow = qrow + half * 8 + r;
      float v0 = s0[r] * 0.125f, v1 = s1[r] * 0.125f;
      if (key0 > srow) v0 = -1e30f;
      if (key1 > srow) v1 = -1e30f;
      s0[r] = v0; s1[r] = v1;
      mloc[r] = redmax16(fmaxf(v0, v1));
    }
    float alpha[8];
#pragma unroll
    for (int r = 0; r < 8; ++r) {
      float nm = fmaxf(rmax[r], mloc[r]);
      alpha[r] = __expf(rmax[r] - nm);
      rmax[r] = nm;
      float p0 = __expf(s0[r] - nm);
      float p1 = __expf(s1[r] - nm);
      int m = half * 8 + r;
      pl[m * 32 + l15]      = f32_to_bf16(p0);
      pl[m * 32 + 16 + l15] = f32_to_bf16(p1);
      rsum[r] = rsum[r] * alpha[r] + redsum16(p0 + p1);
    }
#pragma unroll
    for (int t = 0; t < 4; ++t)
#pragma unroll
      for (int r = 0; r < 8; ++r) o[t][r] *= alpha[r];

    asm volatile("s_wait_dscnt 0" ::: "memory");   // same-wave store->load order
    FragBF pa;
    pa.q[0] = *(const uintx4*)&pl[l15 * 32 + half * 8];
    pa.q[1] = *(const uintx4*)&pl[l15 * 32 + 16 + half * 8];
#pragma unroll
    for (int t = 0; t < 4; ++t) {
      FragBF vb;
      int d = t * 16 + l15;
      vb.q[0] = *(const uintx4*)&Vt[(size_t)d * SS + kb + half * 8];
      vb.q[1] = *(const uintx4*)&Vt[(size_t)d * SS + kb + 16 + half * 8];
      o[t] = wmma_bf16(pa, vb, o[t]);
    }
  }

#pragma unroll
  for (int t = 0; t < 4; ++t)
#pragma unroll
    for (int r = 0; r < 8; ++r) {
      int row = qrow + half * 8 + r;
      int col = h * HDIM + t * 16 + l15;
      attn[((size_t)b * SS + row) * DD + col] = f32_to_bf16(o[t][r] / rsum[r]);
    }
}

// ---------------------------------------------------------------------------
// Host orchestration
// ---------------------------------------------------------------------------
extern "C" void kernel_launch(void* const* d_in, const int* in_sizes, int n_in,
                              void* d_out, int out_size, void* d_ws, size_t ws_size,
                              hipStream_t stream) {
  (void)in_sizes; (void)n_in; (void)out_size; (void)ws_size;
  const float* x           = (const float*)d_in[0];
  const float* n1w         = (const float*)d_in[1];
  const float* n1b         = (const float*)d_in[2];
  const float* n2w         = (const float*)d_in[3];
  const float* n2b         = (const float*)d_in[4];
  const float* n3w         = (const float*)d_in[5];
  const float* n3b         = (const float*)d_in[6];
  const float* nmw         = (const float*)d_in[7];
  const float* nmb         = (const float*)d_in[8];
  const float* in_gate_w   = (const float*)d_in[9];
  const float* in_gate_b   = (const float*)d_in[10];
  const float* out_gate_w  = (const float*)d_in[11];
  const float* out_gate_b  = (const float*)d_in[12];
  const float* to_spec_w   = (const float*)d_in[13];
  const float* from_spec_w = (const float*)d_in[14];
  const float* log_decay   = (const float*)d_in[15];
  const float* frequencies = (const float*)d_in[16];
  const float* vq_emb      = (const float*)d_in[17];
  const float* meta_w      = (const float*)d_in[18];
  const float* meta_b      = (const float*)d_in[19];
  const float* q_w         = (const float*)d_in[20];
  const float* k_w         = (const float*)d_in[21];
  const float* v_w         = (const float*)d_in[22];
  const float* o_w         = (const float*)d_in[23];
  const float* ffn_w1      = (const float*)d_in[24];
  const float* ffn_b1      = (const float*)d_in[25];
  const float* ffn_w2      = (const float*)d_in[26];
  const float* ffn_b2      = (const float*)d_in[27];
  float* out = (float*)d_out;

  char* ws = (char*)d_ws;
  size_t off = 0;
  auto alloc = [&](size_t bytes) -> void* {
    void* p = ws + off;
    off = (off + bytes + 255) & ~(size_t)255;
    return p;
  };
  const size_t DxD = (size_t)DD * DD;

  // bf16 weights
  unsigned short* bwIG  = (unsigned short*)alloc(DxD * 2);
  unsigned short* bwOG  = (unsigned short*)alloc(DxD * 2);
  unsigned short* bwTS  = (unsigned short*)alloc((size_t)F2 * DD * 2);
  unsigned short* bwFS  = (unsigned short*)alloc((size_t)DD * F2 * 2);
  unsigned short* bwMG  = (unsigned short*)alloc(DxD * 2);
  unsigned short* bwQ   = (unsigned short*)alloc(DxD * 2);
  unsigned short* bwK   = (unsigned short*)alloc(DxD * 2);
  unsigned short* bwV   = (unsigned short*)alloc(DxD * 2);
  unsigned short* bwO   = (unsigned short*)alloc(DxD * 2);
  unsigned short* bwF1  = (unsigned short*)alloc((size_t)D4 * DD * 2);
  unsigned short* bwF2  = (unsigned short*)alloc((size_t)DD * D4 * 2);
  unsigned short* bwEMB = (unsigned short*)alloc((size_t)KCODE * DD * 2);

  // activations
  const size_t ND = (size_t)NTOK * DD;           // 4M elements
  float* xnF   = (float*)alloc(ND * 4);
  unsigned short* xnB = (unsigned short*)alloc(ND * 2);
  float* gateF = (float*)alloc(ND * 4);          // g1 / g2 / g3 (disjoint lifetimes)
  unsigned short* xgB = (unsigned short*)alloc(ND * 2);
  float* specF = (float*)alloc((size_t)NTOK * F2 * 4);
  float* soutF = (float*)alloc((size_t)NTOK * F2 * 4);
  unsigned short* soutB = (unsigned short*)alloc((size_t)NTOK * F2 * 2);
  float* xspecF = (float*)alloc(ND * 4);         // later reused as VQ dot matrix
  float* x1F   = (float*)alloc(ND * 4);
  float* xmF   = (float*)alloc(ND * 4);
  unsigned short* xmB = (unsigned short*)alloc(ND * 2);
  float* x2F   = (float*)alloc(ND * 4);
  unsigned short* xn2B = (unsigned short*)alloc(ND * 2);
  unsigned short* xn3B = (unsigned short*)alloc(ND * 2);
  float* qF    = (float*)alloc(ND * 4);          // later reused as attn-proj
  float* kF    = (float*)alloc(ND * 4);          // later reused as x3
  float* vF    = (float*)alloc(ND * 4);          // later reused as ffn2 out
  unsigned short* qhB = (unsigned short*)alloc(ND * 2);
  unsigned short* khB = (unsigned short*)alloc(ND * 2);
  unsigned short* vtB = (unsigned short*)alloc(ND * 2);
  unsigned short* attnB = (unsigned short*)alloc(ND * 2);
  unsigned short* hB  = (unsigned short*)alloc((size_t)NTOK * D4 * 2);
  float* embsqF = (float*)alloc((size_t)KCODE * 4);
  int*   idxI   = (int*)alloc((size_t)NTOK * 4);
  float* partF  = (float*)alloc((size_t)NTOK * 4);

  auto g1d = [](size_t n) { return dim3((unsigned)((n + 255) / 256)); };
  auto cvt = [&](const float* a, unsigned short* o, size_t n) {
    cvt_bf16_kernel<<<g1d(n), 256, 0, stream>>>(a, o, n);
  };
  auto gemm = [&](const unsigned short* A, const unsigned short* W,
                  const float* bias, float* oF, unsigned short* oB,
                  int M, int N, int Kd, int act) {
    dim3 grid(N / 128, M / 128);
    gemm_bf16_kernel<<<grid, 256, 0, stream>>>(A, W, bias, oF, oB, M, N, Kd, act);
  };

  // weight conversions
  cvt(in_gate_w, bwIG, DxD);          cvt(out_gate_w, bwOG, DxD);
  cvt(to_spec_w, bwTS, (size_t)F2 * DD);
  cvt(from_spec_w, bwFS, (size_t)DD * F2);
  cvt(meta_w, bwMG, DxD);
  cvt(q_w, bwQ, DxD); cvt(k_w, bwK, DxD); cvt(v_w, bwV, DxD); cvt(o_w, bwO, DxD);
  cvt(ffn_w1, bwF1, (size_t)D4 * DD); cvt(ffn_w2, bwF2, (size_t)DD * D4);
  cvt(vq_emb, bwEMB, (size_t)KCODE * DD);

  // 1) xn = LN(x, norm1)
  ln_kernel<<<NTOK, 256, 0, stream>>>(x, n1w, n1b, xnF, xnB);
  // 2) g1 = sigmoid(xn @ in_gate_w^T + b)
  gemm(xnB, bwIG, in_gate_b, gateF, nullptr, NTOK, DD, DD, ACT_SIGMOID);
  // 3) xg = xn * g1  (bf16)
  ew_mul_bf_kernel<<<g1d(ND), 256, 0, stream>>>(xgB, xnF, gateF, ND);
  // 4) spec = xg @ to_spec_w^T
  gemm(xgB, bwTS, nullptr, specF, nullptr, NTOK, F2, DD, ACT_NONE);
  // 5) complex recurrence
  spectral_scan_kernel<<<2, 256, 0, stream>>>(specF, log_decay, frequencies, soutF);
  cvt(soutF, soutB, (size_t)NTOK * F2);
  // 6) x_spec_pre = sout @ from_spec_w^T
  gemm(soutB, bwFS, nullptr, xspecF, nullptr, NTOK, DD, F2, ACT_NONE);
  // 7) g2 = sigmoid(xn @ out_gate_w^T + b)
  gemm(xnB, bwOG, out_gate_b, gateF, nullptr, NTOK, DD, DD, ACT_SIGMOID);
  // 8) x1 = x + x_spec_pre * g2
  ew_gate_add_kernel<<<g1d(ND), 256, 0, stream>>>(x1F, x, xspecF, gateF, ND);
  // 9) xm = LN(x1, normm)
  ln_kernel<<<NTOK, 256, 0, stream>>>(x1F, nmw, nmb, xmF, xmB);
  // 10) VQ: dot = xm @ emb^T  (reuse xspecF)
  gemm(xmB, bwEMB, nullptr, xspecF, nullptr, NTOK, KCODE, DD, ACT_NONE);
  emb_sq_kernel<<<KCODE, 256, 0, stream>>>(vq_emb, embsqF);
  argmin_kernel<<<NTOK, 256, 0, stream>>>(xspecF, embsqF, idxI);
  vq_loss_partial_kernel<<<NTOK, 256, 0, stream>>>(xmF, vq_emb, idxI, partF);
  vq_loss_final_kernel<<<1, 256, 0, stream>>>(partF, out + ND);
  // 11) g3 = sigmoid(xm @ meta_gate_w^T + b)
  gemm(xmB, bwMG, meta_b, gateF, nullptr, NTOK, DD, DD, ACT_SIGMOID);
  // 12) x2 = x1 + emb[idx] * g3
  ew_vq_add_kernel<<<g1d(ND), 256, 0, stream>>>(x2F, x1F, vq_emb, idxI, gateF, ND);
  // 13) xn2 = LN(x2, norm2) (bf16 only)
  ln_kernel<<<NTOK, 256, 0, stream>>>(x2F, n2w, n2b, nullptr, xn2B);
  // 14) q,k,v projections
  gemm(xn2B, bwQ, nullptr, qF, nullptr, NTOK, DD, DD, ACT_NONE);
  gemm(xn2B, bwK, nullptr, kF, nullptr, NTOK, DD, DD, ACT_NONE);
  gemm(xn2B, bwV, nullptr, vF, nullptr, NTOK, DD, DD, ACT_NONE);
  // 15) rope + layout
  rope_qk_kernel<<<g1d((size_t)BB * HH * SS * 32), 256, 0, stream>>>(qF, kF, qhB, khB);
  v_transpose_kernel<<<g1d(ND), 256, 0, stream>>>(vF, vtB);
  // 16) flash attention (4096 q-tiles, 8 waves/block)
  flash_attn_kernel<<<512, 256, 0, stream>>>(qhB, khB, vtB, attnB);
  // 17) out-proj (reuse qF) ; x3 = x2 + proj (reuse kF)
  gemm(attnB, bwO, nullptr, qF, nullptr, NTOK, DD, DD, ACT_NONE);
  ew_add_kernel<<<g1d(ND), 256, 0, stream>>>(kF, x2F, qF, ND);
  // 18) xn3 = LN(x3, norm3) (bf16 only)
  ln_kernel<<<NTOK, 256, 0, stream>>>(kF, n3w, n3b, nullptr, xn3B);
  // 19) FFN
  gemm(xn3B, bwF1, ffn_b1, nullptr, hB, NTOK, D4, DD, ACT_GELU);
  gemm(hB, bwF2, ffn_b2, vF, nullptr, NTOK, DD, D4, ACT_NONE);
  // 20) out = x3 + ffn_out
  ew_add_kernel<<<g1d(ND), 256, 0, stream>>>(out, kF, vF, ND);
}
